// SparseMoELayer_75926431858735
// MI455X (gfx1250) — matrix operand order repaired and checked by reading, not compiled, not run on hardware
//
#include <hip/hip_runtime.h>
#include <hip/hip_bf16.h>
#include <math.h>

typedef __attribute__((ext_vector_type(16))) __bf16 v16bf;
typedef __attribute__((ext_vector_type(8)))  __bf16 v8bf;
typedef __attribute__((ext_vector_type(8)))  float  v8f;

#define N_TOK 32768
#define DDIM  256
#define NEXP  8
#define HDIM  1024
#define TM    64      // tokens per block
#define HC    64      // h-chunk (two WMMA K steps for GEMM2)
#define XSTR  264     // lds_x row stride (elems): 132 dwords -> 16 distinct banks
#define HSTR  72      // lds_h row stride (elems): 36 dwords  -> 16 distinct banks

static __device__ __forceinline__ v16bf ld_frag(const __bf16* p0, const __bf16* p1) {
  union { v16bf v; v8bf h[2]; } u;
  u.h[0] = *(const v8bf*)p0;
  u.h[1] = *(const v8bf*)p1;
  return u.v;
}

static __device__ __forceinline__ v8f zero8() {
  v8f v = {0.f, 0.f, 0.f, 0.f, 0.f, 0.f, 0.f, 0.f};
  return v;
}

// ---------------------------------------------------------------------------
// Transpose + fp32->bf16 convert:  in [E][R][C] f32  ->  out [E][C][R] bf16
// ---------------------------------------------------------------------------
__global__ __launch_bounds__(256) void tconv_kernel(const float* __restrict__ in,
                                                    __bf16* __restrict__ out,
                                                    int R, int C) {
  __shared__ float tile[64 * 65];
  const size_t eoff = (size_t)blockIdx.z * R * C;
  in  += eoff;
  out += eoff;
  const int c0 = blockIdx.x * 64, r0 = blockIdx.y * 64;
  const int j = threadIdx.x & 63, i0 = threadIdx.x >> 6;   // 4 rows per pass
  #pragma unroll
  for (int k = 0; k < 16; ++k) {
    const int i = i0 + k * 4;
    tile[i * 65 + j] = in[(size_t)(r0 + i) * C + c0 + j];
  }
  __syncthreads();
  #pragma unroll
  for (int k = 0; k < 16; ++k) {
    const int i = i0 + k * 4;
    out[(size_t)(c0 + i) * R + r0 + j] = (__bf16)tile[j * 65 + i];
  }
}

// ---------------------------------------------------------------------------
// Router: one wave per token. logits = [x, grad] @ Wr + br ; softmax; top-2.
// ---------------------------------------------------------------------------
__global__ __launch_bounds__(256) void router_kernel(const float* __restrict__ x,
                                                     const float* __restrict__ grad,
                                                     const float* __restrict__ Wr,
                                                     const float* __restrict__ br,
                                                     float* __restrict__ probs_out,
                                                     float* __restrict__ cmb) {
  const int lane = threadIdx.x & 31;
  const int wv   = threadIdx.x >> 5;
  const int tok  = blockIdx.x * 8 + wv;

  float acc[NEXP];
  #pragma unroll
  for (int e = 0; e < NEXP; ++e) acc[e] = 0.f;

  const float* xr = x + (size_t)tok * DDIM;
  #pragma unroll
  for (int i = 0; i < DDIM / 32; ++i) {
    const int d = lane + 32 * i;
    const float xv = xr[d];
    const float* wr = Wr + d * NEXP;           // 32B-aligned row of 8 floats
    #pragma unroll
    for (int e = 0; e < NEXP; ++e) acc[e] += xv * wr[e];
  }
  #pragma unroll
  for (int e = 0; e < NEXP; ++e)
    #pragma unroll
    for (int m = 16; m >= 1; m >>= 1) acc[e] += __shfl_xor(acc[e], m, 32);

  const float g = grad[tok];
  float p[NEXP], mx = -1e30f;
  #pragma unroll
  for (int e = 0; e < NEXP; ++e) {
    p[e] = acc[e] + g * Wr[DDIM * NEXP + e] + br[e];
    mx = fmaxf(mx, p[e]);
  }
  float s = 0.f;
  #pragma unroll
  for (int e = 0; e < NEXP; ++e) { p[e] = expf(p[e] - mx); s += p[e]; }
  const float inv = 1.f / s;
  #pragma unroll
  for (int e = 0; e < NEXP; ++e) p[e] *= inv;

  // top-2 (first-index wins ties, matching lax.top_k)
  int i1 = 0;
  #pragma unroll
  for (int e = 1; e < NEXP; ++e) if (p[e] > p[i1]) i1 = e;
  int i2 = -1;
  #pragma unroll
  for (int e = 0; e < NEXP; ++e)
    if (e != i1 && (i2 < 0 || p[e] > p[i2])) i2 = e;

  if (lane == 0) {
    #pragma unroll
    for (int e = 0; e < NEXP; ++e) {
      probs_out[(size_t)tok * NEXP + e] = p[e];
      cmb[(size_t)tok * NEXP + e] = (e == i1 || e == i2) ? p[e] : 0.f;
    }
  }
}

// ---------------------------------------------------------------------------
// Fused MoE: out[t] = sum_e combine[t][e] * (gelu(x W1_e + b1_e) W2_e + b2_e)
// Dense over experts; combine folded into h before GEMM2 so one persistent
// accumulator integrates over all experts and h-chunks.
// ---------------------------------------------------------------------------
__global__ __launch_bounds__(256) void moe_kernel(const float* __restrict__ x,
                                                  const float* __restrict__ b1,
                                                  const float* __restrict__ b2,
                                                  const float* __restrict__ cmb,
                                                  const __bf16* __restrict__ w1t,
                                                  const __bf16* __restrict__ w2t,
                                                  float* __restrict__ out) {
  __shared__ __bf16 lds_x[TM * XSTR];    // x tile, bf16, padded rows
  __shared__ __bf16 lds_h[TM * HSTR];    // gelu'd + combine-scaled h chunk
  __shared__ float  lds_cmb[TM * NEXP];

  const int tid  = threadIdx.x;
  const int lane = tid & 31;
  const int wv   = tid >> 5;
  const int r    = lane & 15;
  const int half = lane >> 4;
  const int tok0 = blockIdx.x * TM;

  for (int i = tid; i < TM * DDIM; i += 256) {
    const int row = i >> 8, col = i & 255;
    lds_x[row * XSTR + col] = (__bf16)x[(size_t)tok0 * DDIM + i];
  }
  for (int i = tid; i < TM * NEXP; i += 256)
    lds_cmb[i] = cmb[(size_t)tok0 * NEXP + i];
  __syncthreads();

  v8f acc[4][2];
  #pragma unroll
  for (int m = 0; m < 4; ++m)
    #pragma unroll
    for (int j = 0; j < 2; ++j) acc[m][j] = zero8();

  const int mt = (wv >> 1) * 16;   // GEMM1: this wave's token-row base

  for (int e = 0; e < NEXP; ++e) {
    const __bf16* w1e = w1t + (size_t)e * HDIM * DDIM;   // [h][d]
    const __bf16* w2e = w2t + (size_t)e * DDIM * HDIM;   // [d][h]
    for (int hc = 0; hc < HDIM; hc += HC) {
      // ---- GEMM1: two 16x16 h-tiles per wave (shared A), K = D = 256
      const int ntc0 = ((wv & 1) * 2 + 0) * 16;   // h-col base, tile 0
      const int ntc1 = ((wv & 1) * 2 + 1) * 16;   // h-col base, tile 1
      v8f hacc0 = zero8(), hacc1 = zero8();
      const __bf16* arow  = lds_x + (mt + r) * XSTR;               // A: row = token
      const __bf16* brow0 = w1e + (size_t)(hc + ntc0 + r) * DDIM;  // B: lane's h column
      const __bf16* brow1 = w1e + (size_t)(hc + ntc1 + r) * DDIM;
      #pragma unroll
      for (int k = 0; k < DDIM; k += 32) {
        v16bf a  = ld_frag(arow + k + half * 8, arow + k + 16 + half * 8);
        v16bf bb0 = ld_frag(brow0 + k + half * 8, brow0 + k + 16 + half * 8);
        v16bf bb1 = ld_frag(brow1 + k + half * 8, brow1 + k + 16 + half * 8);
        hacc0 = __builtin_amdgcn_wmma_f32_16x16x32_bf16(
            false, a, false, bb0, (short)0, hacc0, false, false);
        hacc1 = __builtin_amdgcn_wmma_f32_16x16x32_bf16(
            false, a, false, bb1, (short)0, hacc1, false, false);
      }
      // bias + exact gelu + combine scale -> lds_h (bf16, token-major)
      #pragma unroll
      for (int q = 0; q < 2; ++q) {
        const int ntc = q ? ntc1 : ntc0;
        const v8f& ha = q ? hacc1 : hacc0;
        const float b1v = b1[e * HDIM + hc + ntc + r];
        #pragma unroll
        for (int j = 0; j < 8; ++j) {
          const int m = mt + half * 8 + j;               // C layout: M = half*8 + j
          float v = ha[j] + b1v;
          float gg = 0.5f * v * (1.0f + erff(v * 0.70710678118654752f));
          gg *= lds_cmb[m * NEXP + e];
          lds_h[m * HSTR + ntc + r] = (__bf16)gg;
        }
      }
      __syncthreads();
      // ---- GEMM2: y tiles (4 m-tiles x this wave's 2 n-tiles), K = HC = 64
      #pragma unroll
      for (int j = 0; j < 2; ++j) {
        const int nb = wv * 32 + j * 16;
        #pragma unroll
        for (int kk = 0; kk < 2; ++kk) {
          const __bf16* b2row =
              w2e + (size_t)(nb + r) * HDIM + hc + kk * 32;
          v16bf b = ld_frag(b2row + half * 8, b2row + 16 + half * 8);
          #pragma unroll
          for (int m = 0; m < 4; ++m) {
            const __bf16* ar = lds_h + (m * 16 + r) * HSTR + kk * 32;
            v16bf a = ld_frag(ar + half * 8, ar + 16 + half * 8);
            acc[m][j] = __builtin_amdgcn_wmma_f32_16x16x32_bf16(
                false, a, false, b, (short)0, acc[m][j], false, false);
          }
        }
      }
      __syncthreads();
    }
  }

  // epilogue: + sum_e combine[t][e] * b2[e][n], store fp32
  #pragma unroll
  for (int j = 0; j < 2; ++j) {
    const int n = wv * 32 + j * 16 + r;
    float b2v[NEXP];
    #pragma unroll
    for (int e = 0; e < NEXP; ++e) b2v[e] = b2[e * DDIM + n];
    #pragma unroll
    for (int m = 0; m < 4; ++m) {
      #pragma unroll
      for (int jj = 0; jj < 8; ++jj) {
        const int tr = m * 16 + half * 8 + jj;
        float bias = 0.f;
        #pragma unroll
        for (int e = 0; e < NEXP; ++e) bias += lds_cmb[tr * NEXP + e] * b2v[e];
        out[(size_t)(tok0 + tr) * DDIM + n] = acc[m][j][jj] + bias;
      }
    }
  }
}

// ---------------------------------------------------------------------------
extern "C" void kernel_launch(void* const* d_in, const int* in_sizes, int n_in,
                              void* d_out, int out_size, void* d_ws, size_t ws_size,
                              hipStream_t stream) {
  (void)in_sizes; (void)n_in; (void)out_size; (void)ws_size;
  const float* x    = (const float*)d_in[0];
  const float* grad = (const float*)d_in[1];
  const float* Wr   = (const float*)d_in[2];
  const float* br   = (const float*)d_in[3];
  const float* W1   = (const float*)d_in[4];
  const float* b1   = (const float*)d_in[5];
  const float* W2   = (const float*)d_in[6];
  const float* b2   = (const float*)d_in[7];

  float* out   = (float*)d_out;                          // [N, D]
  float* probs = out + (size_t)N_TOK * DDIM;             // [N, E]

  char* ws = (char*)d_ws;
  const size_t cmb_bytes = (size_t)N_TOK * NEXP * sizeof(float);   // 1 MB
  const size_t wt_bytes  = (size_t)NEXP * HDIM * DDIM * 2;         // 4 MB each
  float*  cmb = (float*)ws;
  __bf16* w1t = (__bf16*)(ws + cmb_bytes);               // [E][H][D] bf16
  __bf16* w2t = (__bf16*)(ws + cmb_bytes + wt_bytes);    // [E][D][H] bf16

  // W1 [E][D][H] -> W1t [E][H][D] ;  W2 [E][H][D] -> W2t [E][D][H]
  tconv_kernel<<<dim3(HDIM / 64, DDIM / 64, NEXP), 256, 0, stream>>>(W1, w1t, DDIM, HDIM);
  tconv_kernel<<<dim3(DDIM / 64, HDIM / 64, NEXP), 256, 0, stream>>>(W2, w2t, HDIM, DDIM);

  router_kernel<<<N_TOK / 8, 256, 0, stream>>>(x, grad, Wr, br, probs, cmb);

  moe_kernel<<<N_TOK / TM, 256, 0, stream>>>(x, b1, b2, cmb, w1t, w2t, out);
}
